// MoE_41394894799020
// MI455X (gfx1250) — matrix operand order repaired and checked by reading, not compiled, not run on hardware
//
#include <hip/hip_runtime.h>
#include <hip/hip_fp16.h>

typedef __attribute__((ext_vector_type(16))) _Float16 v16h;
typedef __attribute__((ext_vector_type(8)))  float    v8f;

union F16Frag { v16h v; unsigned u[8]; };

#if __has_builtin(__builtin_amdgcn_global_load_async_to_lds_b128)
#define HAS_ASYNC_LDS 1
#endif

#ifdef HAS_ASYNC_LDS
// Builtin signature (from hipcc diagnostic): param0 = global  'v4i*' (AS1,
// printed as '__device__'), param1 = LDS 'v4i*' (AS3), then imm offset, imm cpol.
typedef int v4i_b128 __attribute__((vector_size(16)));
typedef __attribute__((address_space(1))) v4i_b128* g_v4i_ptr;
typedef __attribute__((address_space(3))) v4i_b128* l_v4i_ptr;

__device__ __forceinline__ void async_copy_b128(const _Float16* src, _Float16* dst)
{
    __builtin_amdgcn_global_load_async_to_lds_b128(
        (g_v4i_ptr)src, (l_v4i_ptr)dst, 0 /*offset*/, 0 /*cpol*/);
}
__device__ __forceinline__ void async_wait_all()
{
#if __has_builtin(__builtin_amdgcn_s_wait_asynccnt)
    __builtin_amdgcn_s_wait_asynccnt(0);
#else
    asm volatile("s_wait_asynccnt 0x0" ::: "memory");
#endif
}
#endif

// ---------------------------------------------------------------------------
// Fold conv bias + eval-mode BN into per-channel scale/shift:
//   y = (acc + b)*g/sqrt(v+eps) + (bt - m*g/sqrt(v+eps)) = acc*s + ((b-m)*s+bt)
// ---------------------------------------------------------------------------
__global__ void k_bn(const float* __restrict__ g, const float* __restrict__ bt,
                     const float* __restrict__ rm, const float* __restrict__ rv,
                     const float* __restrict__ bias, float* __restrict__ s,
                     float* __restrict__ t, int C)
{
    int i = threadIdx.x;
    if (i < C) {
        float sv = g[i] * rsqrtf(rv[i] + 1e-5f);
        s[i] = sv;
        t[i] = (bias[i] - rm[i]) * sv + bt[i];
    }
}

// ---------------------------------------------------------------------------
// Pack conv weights f32 OIHW -> f16 [N][K], K = (ky*3+kx)*Cin + c
// ---------------------------------------------------------------------------
__global__ void k_packw(const float* __restrict__ w, _Float16* __restrict__ wp,
                        int Cin, int Cout)
{
    long gid = (long)blockIdx.x * 256 + threadIdx.x;
    long total = (long)Cout * Cin * 9;
    if (gid >= total) return;
    int Kl = Cin * 9;
    int n  = (int)(gid / Kl);
    int kk = (int)(gid % Kl);
    int r3 = kk / Cin;             // ky*3+kx
    int c  = kk - r3 * Cin;
    wp[gid] = (_Float16)w[(long)(n * Cin + c) * 9 + r3];
}

// ---------------------------------------------------------------------------
// Layer 0: direct conv 3->32, 32x32, SAME + fused BN/ReLU, out f16 NHWC.
// ---------------------------------------------------------------------------
__global__ __launch_bounds__(256) void k_conv0(
    const float* __restrict__ xin, const float* __restrict__ w,
    const float* __restrict__ s, const float* __restrict__ t,
    _Float16* __restrict__ out)
{
    long o = (long)blockIdx.x * 256 + threadIdx.x;   // B*32*32*32
    if (o >= (long)512 * 32 * 32 * 32) return;
    int co = (int)(o & 31);
    int x  = (int)((o >> 5)  & 31);
    int y  = (int)((o >> 10) & 31);
    int b  = (int)(o >> 15);
    float acc = 0.f;
    #pragma unroll
    for (int ci = 0; ci < 3; ++ci)
        #pragma unroll
        for (int ky = 0; ky < 3; ++ky) {
            int yy = y + ky - 1;
            if ((unsigned)yy >= 32u) continue;
            #pragma unroll
            for (int kx = 0; kx < 3; ++kx) {
                int xx = x + kx - 1;
                if ((unsigned)xx >= 32u) continue;
                acc += xin[(((long)b * 3 + ci) * 32 + yy) * 32 + xx] *
                       w[((co * 3 + ci) * 3 + ky) * 3 + kx];
            }
        }
    float val = acc * s[co] + t[co];
    out[(((long)b * 32 + y) * 32 + x) * 32 + co] = (_Float16)(val > 0.f ? val : 0.f);
}

// ---------------------------------------------------------------------------
// Implicit-GEMM conv via V_WMMA_F32_16X16X32_F16, double-buffered LDS
// pipeline with async global->LDS staging of the weight tile.
//   M = B*H*W, N = Cout, K = Cin*9 (Cin mult of 32).
//   Block: 256 threads = 8 waves; tile M=128 x N=64; wave tile 32x32 (2x2 WMMA).
// ---------------------------------------------------------------------------
__global__ __launch_bounds__(256) void k_conv_wmma(
    const _Float16* __restrict__ in, const _Float16* __restrict__ wp,
    const float* __restrict__ scale, const float* __restrict__ shift,
    _Float16* __restrict__ out, int H, int W, int Cin, int Cout)
{
    const int Kdim = Cin * 9;
    const int nK   = Kdim >> 5;
    const int HW   = H * W;
    const long Mtot = (long)512 * HW;

    __shared__ _Float16 As[2][128][48];   // double-buffered: 128 pixels x 32 K
    __shared__ _Float16 Bs[2][64][48];    // double-buffered: 64 N x 32 K ([n][k])

    const int tid    = threadIdx.x;
    const long mbase = (long)blockIdx.x * 128;
    const int nbase  = blockIdx.y * 64;

    // A-staging coords: 2 threads per pixel, each 16 channels (32B = 2x b128)
    const int p  = tid >> 1;
    const int ah = tid & 1;
    const long m = mbase + p;
    const int b  = (int)(m / HW);
    const int rem = (int)(m % HW);
    const int y  = rem / W;
    const int x  = rem % W;

    // B-staging coords: 4 threads per output channel row, each 8 K (16B)
    const int bcol = tid >> 2;
    const int bseg = tid & 3;

    const int wid   = tid >> 5;
    const int lane  = tid & 31;
    const int rl    = lane & 15;
    const int hi    = lane >> 4;
    const int waveM = (wid >> 1) * 32;
    const int waveN = (wid & 1) * 32;

    auto stageA = [&](int bufi, int ks) {
        const int r3 = ks / Cin;            // tap index ky*3+kx
        const int cb = ks - r3 * Cin;       // channel base (mult of 32)
        const int ky = r3 / 3, kx = r3 - ky * 3;
        const int yy = y + ky - 1, xx = x + kx - 1;
        uint4 v0 = {0u, 0u, 0u, 0u}, v1 = {0u, 0u, 0u, 0u};
        if ((unsigned)yy < (unsigned)H && (unsigned)xx < (unsigned)W) {
            const _Float16* src = in + (((long)b * H + yy) * W + xx) * Cin + cb + ah * 16;
            v0 = *(const uint4*)(src);
            v1 = *(const uint4*)(src + 8);
        }
        *(uint4*)&As[bufi][p][ah * 16]     = v0;
        *(uint4*)&As[bufi][p][ah * 16 + 8] = v1;
    };
    auto stageB = [&](int bufi, int ks) {
        const _Float16* src = wp + (long)(nbase + bcol) * Kdim + ks + bseg * 8;
        _Float16* dst = &Bs[bufi][bcol][bseg * 8];
#ifdef HAS_ASYNC_LDS
        async_copy_b128(src, dst);          // -> global_load_async_to_lds_b128
#else
        *(uint4*)dst = *(const uint4*)src;
#endif
    };

    v8f c00 = {}, c01 = {}, c10 = {}, c11 = {};

    // prologue: stage chunk 0 into buffer 0
    stageA(0, 0);
    stageB(0, 0);
#ifdef HAS_ASYNC_LDS
    async_wait_all();
#endif
    __syncthreads();

    for (int it = 0; it < nK; ++it) {
        const int cur = it & 1;
        // stage next chunk into the other buffer while computing this one
        if (it + 1 < nK) {
            stageA(cur ^ 1, (it + 1) << 5);
            stageB(cur ^ 1, (it + 1) << 5);
        }

        // ---- fragments per CDNA5 16-bit A(16x32)/B(32x16) layouts ----
        F16Frag a0, a1, b0, b1;
        #pragma unroll
        for (int v = 0; v < 8; ++v) {
            // A: VGPR v<4 -> K=2v(+1); v>=4 -> K=16+2(v-4); lanes 16-31: +8
            const int ka = (v < 4 ? 2 * v : 16 + 2 * (v - 4)) + hi * 8;
            a0.u[v] = *(const unsigned*)&As[cur][waveM + rl][ka];
            a1.u[v] = *(const unsigned*)&As[cur][waveM + 16 + rl][ka];
            // B: VGPR v<4 -> K=2v(+1); v>=4 -> K=16+2(v-4); lanes 16-31: +8
            const int kb = (v < 4 ? 2 * v : 16 + 2 * (v - 4)) + hi * 8;
            b0.u[v] = *(const unsigned*)&Bs[cur][waveN + rl][kb];
            b1.u[v] = *(const unsigned*)&Bs[cur][waveN + 16 + rl][kb];
        }
        c00 = __builtin_amdgcn_wmma_f32_16x16x32_f16(false, a0.v, false, b0.v, (short)0, c00, false, false);
        c01 = __builtin_amdgcn_wmma_f32_16x16x32_f16(false, a0.v, false, b1.v, (short)0, c01, false, false);
        c10 = __builtin_amdgcn_wmma_f32_16x16x32_f16(false, a1.v, false, b0.v, (short)0, c10, false, false);
        c11 = __builtin_amdgcn_wmma_f32_16x16x32_f16(false, a1.v, false, b1.v, (short)0, c11, false, false);

#ifdef HAS_ASYNC_LDS
        async_wait_all();                   // drain ASYNCcnt before handoff
#endif
        __syncthreads();
    }

    // ---- epilogue: fused BN + ReLU, f16 NHWC store ----
    #pragma unroll
    for (int j = 0; j < 2; ++j) {
        const int gn = nbase + waveN + j * 16 + rl;
        const float sN = scale[gn], tN = shift[gn];
        #pragma unroll
        for (int i = 0; i < 2; ++i) {
            v8f acc = (i == 0) ? (j == 0 ? c00 : c01) : (j == 0 ? c10 : c11);
            #pragma unroll
            for (int v = 0; v < 8; ++v) {
                const long gm = mbase + waveM + i * 16 + hi * 8 + v;  // D: M = v + 8*hi
                if (gm < Mtot) {
                    const int rr = (int)(gm % HW);
                    const long bb = gm / HW;
                    float val = acc[v] * sN + tN;
                    val = val > 0.f ? val : 0.f;
                    out[(bb * HW + rr) * Cout + gn] = (_Float16)val;
                }
            }
        }
    }
}

// ---------------------------------------------------------------------------
// 2x2 maxpool, f16 NHWC.
// ---------------------------------------------------------------------------
__global__ void k_maxpool(const _Float16* __restrict__ in, _Float16* __restrict__ out,
                          int Hi, int Wi, int C, long total)
{
    long o = (long)blockIdx.x * 256 + threadIdx.x;
    if (o >= total) return;
    int c = (int)(o % C);
    long q = o / C;
    int Wo = Wi >> 1, Ho = Hi >> 1;
    int xo = (int)(q % Wo); q /= Wo;
    int yo = (int)(q % Ho);
    int b  = (int)(q / Ho);
    long base = (((long)b * Hi + 2 * yo) * Wi + 2 * xo);
    float a0 = (float)in[base * C + c];
    float a1 = (float)in[(base + 1) * C + c];
    float a2 = (float)in[(base + Wi) * C + c];
    float a3 = (float)in[(base + Wi + 1) * C + c];
    out[o] = (_Float16)fmaxf(fmaxf(a0, a1), fmaxf(a2, a3));
}

// ---------------------------------------------------------------------------
// Global average pool over 8x8, f16 [512,8,8,256] -> f32 feats [512,256]
// ---------------------------------------------------------------------------
__global__ void k_avgpool(const _Float16* __restrict__ in, float* __restrict__ feats)
{
    int gid = blockIdx.x * 256 + threadIdx.x;   // 512*256
    int c = gid & 255, b = gid >> 8;
    const _Float16* p = in + (long)b * 64 * 256 + c;
    float acc = 0.f;
    #pragma unroll 8
    for (int q = 0; q < 64; ++q) acc += (float)p[q * 256];
    feats[gid] = acc * (1.0f / 64.0f);
}

// ---------------------------------------------------------------------------
// Per (b,e): 10 logits, log-softmax, conf = -entropy = sum p*logp
// ---------------------------------------------------------------------------
__global__ __launch_bounds__(256) void k_classify(
    const float* __restrict__ feats, const float* __restrict__ cw,
    const float* __restrict__ cb, float* __restrict__ logits,
    float* __restrict__ confo)
{
    int gid = blockIdx.x * 256 + threadIdx.x;   // 512*8
    int b = gid >> 3, e = gid & 7;
    const float* f = feats + (long)b * 256;
    float l[10];
    #pragma unroll
    for (int d = 0; d < 10; ++d) {
        const float* wrow = cw + (long)(e * 10 + d) * 256;
        float acc = cb[e * 10 + d];
        for (int c = 0; c < 256; ++c) acc += f[c] * wrow[c];
        l[d] = acc;
        logits[(long)gid * 10 + d] = acc;
    }
    float mx = l[0];
    #pragma unroll
    for (int d = 1; d < 10; ++d) mx = fmaxf(mx, l[d]);
    float se = 0.f;
    #pragma unroll
    for (int d = 0; d < 10; ++d) se += expf(l[d] - mx);
    float lse = mx + logf(se);
    float cf = 0.f;
    #pragma unroll
    for (int d = 0; d < 10; ++d) { float lp = l[d] - lse; cf += expf(lp) * lp; }
    confo[gid] = cf;
}

// ---------------------------------------------------------------------------
// Greedy capacity-limited routing, 4 cycles. One workgroup:
//   bitonic sort of 4096 keys in LDS (desc score, tie asc idx == stable
//   argsort(-score)), then single-lane sequential commit scan.
// ---------------------------------------------------------------------------
__global__ __launch_bounds__(1024) void k_route(const float* __restrict__ conf,
                                                int* __restrict__ Dint,
                                                float* __restrict__ Dout)
{
    __shared__ unsigned long long keys[4096];
    __shared__ float cur[4096];
    __shared__ unsigned char Df[4096];
    __shared__ int ec[8];
    __shared__ int sc[512];
    __shared__ int assigned;
    const int tid = threadIdx.x;

    for (int i = tid; i < 4096; i += 1024) { cur[i] = conf[i]; Df[i] = 0; }
    for (int i = tid; i < 512; i += 1024) sc[i] = 0;
    if (tid < 8) ec[tid] = 0;
    if (tid == 0) assigned = 0;
    __syncthreads();

    for (int cycle = 0; cycle < 4; ++cycle) {
        for (int i = tid; i < 4096; i += 1024) {
            unsigned bits = __float_as_uint(cur[i]);
            unsigned asc  = (bits & 0x80000000u) ? ~bits : (bits | 0x80000000u);
            unsigned desc = ~asc;   // ascending sort on desc == descending score
            keys[i] = ((unsigned long long)desc << 32) | (unsigned)i;
        }
        __syncthreads();
        for (int k2 = 2; k2 <= 4096; k2 <<= 1)
            for (int j = k2 >> 1; j > 0; j >>= 1) {
                for (int i = tid; i < 4096; i += 1024) {
                    int ixj = i ^ j;
                    if (ixj > i) {
                        bool up = ((i & k2) == 0);
                        unsigned long long ka = keys[i], kb = keys[ixj];
                        if ((ka > kb) == up) { keys[i] = kb; keys[ixj] = ka; }
                    }
                }
                __syncthreads();
            }
        if (tid == 0) {
            int cyc = 0, asg = assigned;
            for (int s2 = 0; s2 < 4096; ++s2) {
                int idx = (int)(keys[s2] & 0xFFFFFFFFull);
                int bb = idx >> 3, jj = idx & 7;
                if (!Df[idx] && ec[jj] < 160 && sc[bb] < 2 && cyc < 256 && asg < 1024) {
                    Df[idx] = 1; ec[jj]++; sc[bb]++; asg++; cyc++;
                }
            }
            assigned = asg;
        }
        __syncthreads();
        for (int i = tid; i < 4096; i += 1024)
            cur[i] = conf[i] * (1.0f - (float)ec[i & 7] * (1.0f / 160.0f));
        __syncthreads();
    }
    for (int i = tid; i < 4096; i += 1024) {
        Dint[i] = Df[i];
        Dout[i] = Df[i] ? 1.0f : 0.0f;
    }
}

// ---------------------------------------------------------------------------
// final[b,c] = sum_e D*conf*logits / max(#D, 1)
// ---------------------------------------------------------------------------
__global__ void k_final(const float* __restrict__ logits, const float* __restrict__ conf,
                        const int* __restrict__ Dint, float* __restrict__ out)
{
    int gid = blockIdx.x * 256 + threadIdx.x;
    if (gid >= 5120) return;
    int b = gid / 10, d = gid % 10;
    float acc = 0.f; int cnt = 0;
    #pragma unroll
    for (int e = 0; e < 8; ++e) {
        if (Dint[b * 8 + e]) {
            acc += conf[b * 8 + e] * logits[((long)b * 8 + e) * 10 + d];
            cnt++;
        }
    }
    out[gid] = acc / (cnt > 0 ? (float)cnt : 1.0f);
}

// ---------------------------------------------------------------------------
extern "C" void kernel_launch(void* const* d_in, const int* in_sizes, int n_in,
                              void* d_out, int out_size, void* d_ws, size_t ws_size,
                              hipStream_t stream)
{
    const float* x = (const float*)d_in[0];
    const float *w[4], *bia[4], *g[4], *bt[4], *rm[4], *rv[4];
    for (int i = 0; i < 4; ++i) {
        w[i]   = (const float*)d_in[1 + i * 6 + 0];
        bia[i] = (const float*)d_in[1 + i * 6 + 1];
        g[i]   = (const float*)d_in[1 + i * 6 + 2];
        bt[i]  = (const float*)d_in[1 + i * 6 + 3];
        rm[i]  = (const float*)d_in[1 + i * 6 + 4];
        rv[i]  = (const float*)d_in[1 + i * 6 + 5];
    }
    const float* cls_w = (const float*)d_in[25];
    const float* cls_b = (const float*)d_in[26];

    char* base = (char*)d_ws;
    size_t off = 0;
    auto alloc = [&](size_t bytes) -> void* {
        off = (off + 255) & ~(size_t)255;
        void* p = base + off;
        off += bytes;
        return p;
    };

    const int Couts[4] = {32, 64, 128, 256};
    float *s[4], *tt[4];
    for (int i = 0; i < 4; ++i) {
        s[i]  = (float*)alloc((size_t)Couts[i] * 4);
        tt[i] = (float*)alloc((size_t)Couts[i] * 4);
    }
    _Float16* wp1 = (_Float16*)alloc((size_t)64  * 288  * 2);
    _Float16* wp2 = (_Float16*)alloc((size_t)128 * 576  * 2);
    _Float16* wp3 = (_Float16*)alloc((size_t)256 * 1152 * 2);
    _Float16* buf0 = (_Float16*)alloc((size_t)16777216 * 2);  // a0 / a2
    _Float16* buf1 = (_Float16*)alloc((size_t)33554432 * 2);  // a1 / a3
    _Float16* buf2 = (_Float16*)alloc((size_t)8388608  * 2);  // a1p / a3p
    float* feats  = (float*)alloc((size_t)131072 * 4);
    float* logits = (float*)alloc((size_t)40960 * 4);
    int*   Dint   = (int*)alloc((size_t)4096 * 4);
    (void)ws_size; (void)in_sizes; (void)n_in; (void)out_size;

    float* out_final = (float*)d_out;        // [512,10]
    float* out_conf  = out_final + 5120;     // [512,8]
    float* out_D     = out_conf + 4096;      // [512,8]

    for (int i = 0; i < 4; ++i)
        k_bn<<<1, 256, 0, stream>>>(g[i], bt[i], rm[i], rv[i], bia[i], s[i], tt[i], Couts[i]);

    k_packw<<<72,   256, 0, stream>>>(w[1], wp1, 32, 64);
    k_packw<<<288,  256, 0, stream>>>(w[2], wp2, 64, 128);
    k_packw<<<1152, 256, 0, stream>>>(w[3], wp3, 128, 256);

    k_conv0<<<65536, 256, 0, stream>>>(x, w[0], s[0], tt[0], buf0);

    k_conv_wmma<<<dim3(4096, 1), 256, 0, stream>>>(buf0, wp1, s[1], tt[1], buf1, 32, 32, 32, 64);
    k_maxpool<<<32768, 256, 0, stream>>>(buf1, buf2, 32, 32, 64, (long)8388608);
    k_conv_wmma<<<dim3(1024, 2), 256, 0, stream>>>(buf2, wp2, s[2], tt[2], buf0, 16, 16, 64, 128);
    k_conv_wmma<<<dim3(1024, 4), 256, 0, stream>>>(buf0, wp3, s[3], tt[3], buf1, 16, 16, 128, 256);
    k_maxpool<<<32768, 256, 0, stream>>>(buf1, buf2, 16, 16, 256, (long)8388608);
    k_avgpool<<<512, 256, 0, stream>>>(buf2, feats);
    k_classify<<<16, 256, 0, stream>>>(feats, cls_w, cls_b, logits, out_conf);
    k_route<<<1, 1024, 0, stream>>>(out_conf, Dint, out_D);
    k_final<<<20, 256, 0, stream>>>(logits, out_conf, Dint, out_final);
}
